// PCRBase_85074712199222
// MI455X (gfx1250) — matrix operand order repaired and checked by reading, not compiled, not run on hardware
//
#include <hip/hip_runtime.h>
#include <hip/hip_bf16.h>

typedef __attribute__((ext_vector_type(16))) _Float16 v16h;
typedef __attribute__((ext_vector_type(8)))  float    v8f;
typedef __attribute__((ext_vector_type(2)))  float    v2f;

#define B_  16
#define N_  1024
#define F_  16
#define E_  12
#define H_  4
#define DH_ 3

// ---------------------------------------------------------------------------
// Kernel 1: per-batch mean over N, center x/y, project F=16 -> E=12 with W_in.
// ---------------------------------------------------------------------------
__global__ void k_center_project(const float* __restrict__ x, const float* __restrict__ y,
                                 const float* __restrict__ Win, const float* __restrict__ bin,
                                 float* __restrict__ x3c, float* __restrict__ xi,
                                 float* __restrict__ yi, float* __restrict__ ymean3)
{
    __shared__ float smx[F_], smy[F_], sW[E_ * F_], sb[E_];
    const int b = blockIdx.x, t = threadIdx.x;
    if (t < F_) { smx[t] = 0.f; smy[t] = 0.f; }
    for (int i = t; i < E_ * F_; i += blockDim.x) sW[i] = Win[i];
    if (t < E_) sb[t] = bin[t];
    __syncthreads();

    float ax[F_], ay[F_];
#pragma unroll
    for (int f = 0; f < F_; ++f) { ax[f] = 0.f; ay[f] = 0.f; }
    for (int n = t; n < N_; n += 256) {
        const float* xr = &x[(b * N_ + n) * F_];
        const float* yr = &y[(b * N_ + n) * F_];
#pragma unroll
        for (int f = 0; f < F_; ++f) { ax[f] += xr[f]; ay[f] += yr[f]; }
    }
#pragma unroll
    for (int f = 0; f < F_; ++f) { atomicAdd(&smx[f], ax[f]); atomicAdd(&smy[f], ay[f]); }
    __syncthreads();

    float mx[F_], my[F_];
#pragma unroll
    for (int f = 0; f < F_; ++f) { mx[f] = smx[f] * (1.f / N_); my[f] = smy[f] * (1.f / N_); }
    if (t < 3) ymean3[b * 3 + t] = my[t];

    for (int n = t; n < N_; n += 256) {
        const float* xr = &x[(b * N_ + n) * F_];
        const float* yr = &y[(b * N_ + n) * F_];
        float xc[F_], yc[F_];
#pragma unroll
        for (int f = 0; f < F_; ++f) { xc[f] = xr[f] - mx[f]; yc[f] = yr[f] - my[f]; }
#pragma unroll
        for (int j = 0; j < 3; ++j) x3c[(b * N_ + n) * 3 + j] = xc[j];
#pragma unroll
        for (int e = 0; e < E_; ++e) {
            float sx = sb[e], sy = sb[e];
#pragma unroll
            for (int f = 0; f < F_; ++f) { sx += xc[f] * sW[e * F_ + f]; sy += yc[f] * sW[e * F_ + f]; }
            xi[(b * N_ + n) * E_ + e] = sx;
            yi[(b * N_ + n) * E_ + e] = sy;
        }
    }
}

// ---------------------------------------------------------------------------
// Kernel 2: QKV projection.  src (B*N,12) -> q/k (B,H,N,DH), v TRANSPOSED as
// (B,H,DH,N) so the attention kernel's V tile loads are contiguous.
// ---------------------------------------------------------------------------
__global__ void k_qkv(const float* __restrict__ src, const float* __restrict__ W,
                      const float* __restrict__ bias,
                      float* __restrict__ qo, float* __restrict__ ko, float* __restrict__ vo)
{
    __shared__ float sW[3 * E_ * E_], sb[3 * E_];
    const int t = threadIdx.x;
    for (int i = t; i < 3 * E_ * E_; i += blockDim.x) sW[i] = W[i];
    if (t < 3 * E_) sb[t] = bias[t];
    __syncthreads();

    const int rid = blockIdx.x * blockDim.x + t;   // = b*N + n
    if (rid >= B_ * N_) return;
    const int b = rid / N_, n = rid % N_;

    float in[E_];
#pragma unroll
    for (int e = 0; e < E_; ++e) in[e] = src[rid * E_ + e];

    float o[3 * E_];
#pragma unroll
    for (int r = 0; r < 3 * E_; ++r) {
        float s = sb[r];
#pragma unroll
        for (int e = 0; e < E_; ++e) s += in[e] * sW[r * E_ + e];
        o[r] = s;
    }
#pragma unroll
    for (int h = 0; h < H_; ++h)
#pragma unroll
        for (int d = 0; d < DH_; ++d) {
            const int idx = ((b * H_ + h) * N_ + n) * DH_ + d;
            if (qo) qo[idx] = o[h * DH_ + d];
            if (ko) ko[idx] = o[E_ + h * DH_ + d];
            if (vo) vo[((b * H_ + h) * DH_ + d) * N_ + n] = o[2 * E_ + h * DH_ + d];
        }
}

// ---------------------------------------------------------------------------
// Kernel 3: attention, one wave32 per 16-query tile of one (b,h) pair.
//   S tile = V_WMMA_F32_16X16X4_F32 (K=4 fits d_h=3 exactly, full f32)
//   softmax without max-shift (scores tiny; exp-safe); row sums reduced once.
//   O += V_WMMA_F32_16X16X32_F16(P, V), P moved C->A layout via LDS.
//   Operand builds are branchless (cndmask selects, no exec diamonds).
// ---------------------------------------------------------------------------
__global__ void k_flash(const float* __restrict__ q, const float* __restrict__ k,
                        const float* __restrict__ v, float* __restrict__ out)
{
    __shared__ _Float16 pbuf[8][16][16];   // per-wave 16x16 P tile
    const int lane  = threadIdx.x & 31;
    const int wave  = threadIdx.x >> 5;
    const int b     = blockIdx.z, h = blockIdx.y;
    const int bh    = b * H_ + h;
    const int qbase = (blockIdx.x * 8 + wave) * 16;
    const int n     = lane & 15;       // C-layout column / matrix row index
    const int hi    = lane >> 4;       // C-layout row-half
    const bool lo   = (lane < 16);
    const float scale = 0.5773502691896258f;   // 1/sqrt(dh)

    // Q as f32 A-matrix (16x4): lanes 0-15 hold K=0,1; lanes 16-31 hold K=2,3.
    // Branchless: both half-waves load the same 3-float row, then select.
    v2f aq;
    {
        const float* qr = &q[(bh * N_ + qbase + n) * DH_];
        const float q0 = qr[0], q1 = qr[1], q2 = qr[2];
        aq[0] = (lo ? q0 : q2) * scale;
        aq[1] = lo ? q1 * scale : 0.f;
    }

    float lsum[8];
    v8f o;
#pragma unroll
    for (int r = 0; r < 8; ++r) { lsum[r] = 0.f; o[r] = 0.f; }

    for (int kt = 0; kt < N_ / 16; ++kt) {
        // K as f32 B-matrix (4x16), branchless build from one b96 load
        v2f bk;
        {
            const float* kr = &k[(bh * N_ + kt * 16 + n) * DH_];
            const float k0 = kr[0], k1 = kr[1], k2 = kr[2];
            bk[0] = lo ? k0 : k2;
            bk[1] = lo ? k1 : 0.f;
            __builtin_prefetch(kr + 16 * DH_, 0, 1);   // next key tile (L2-resident anyway)
        }
        v8f zero;
#pragma unroll
        for (int r = 0; r < 8; ++r) zero[r] = 0.f;
        v8f s = __builtin_amdgcn_wmma_f32_16x16x4_f32(false, aq, false, bk,
                                                      (short)0, zero, false, false);
        // un-shifted exp (scores tiny; softmax is shift-invariant)
        float p[8];
#pragma unroll
        for (int r = 0; r < 8; ++r) { p[r] = __expf(s[r]); lsum[r] += p[r]; }

        // C-layout -> A-layout transpose of P through LDS (per-wave region)
#pragma unroll
        for (int r = 0; r < 8; ++r) pbuf[wave][r + 8 * hi][n] = (_Float16)p[r];

        v16h ap;
#pragma unroll
        for (int i = 0; i < 16; ++i) ap[i] = (_Float16)0.f;
        {
            const int m  = lo ? lane : lane - 16;
#pragma unroll
            for (int i = 0; i < 16; ++i) ap[i] = lo ? pbuf[wave][m][i] : (_Float16)0.f;
        }
        // V as f16 B-matrix (16pad32 x 16): lanes (=output d) 0..2 read 16
        // CONTIGUOUS floats from the transposed (B,H,DH,N) V layout.
        v16h bv;
#pragma unroll
        for (int i = 0; i < 16; ++i) bv[i] = (_Float16)0.f;
        if (lane < DH_) {
            const float* vr = &v[(bh * DH_ + lane) * N_ + kt * 16];
#pragma unroll
            for (int i = 0; i < 16; ++i) bv[i] = (_Float16)vr[i];
            __builtin_prefetch(vr + 16, 0, 1);
        }
        o = __builtin_amdgcn_wmma_f32_16x16x32_f16(false, ap, false, bv,
                                                   (short)0, o, false, false);
    }

    // one final 16-lane reduction of the row sums
#pragma unroll
    for (int r = 0; r < 8; ++r) {
        float s = lsum[r];
        s += __shfl_xor(s, 1);
        s += __shfl_xor(s, 2);
        s += __shfl_xor(s, 4);
        s += __shfl_xor(s, 8);
        lsum[r] = s;
    }

    if (n < DH_) {
#pragma unroll
        for (int r = 0; r < 8; ++r) {
            const int m = r + 8 * hi;
            out[(b * N_ + qbase + m) * E_ + h * DH_ + n] = o[r] / lsum[r];
        }
    }
}

// ---------------------------------------------------------------------------
// Kernel 4: 12->12 output projection (Wo^T + bo).
// ---------------------------------------------------------------------------
__global__ void k_proj12(const float* __restrict__ in, const float* __restrict__ W,
                         const float* __restrict__ bias, float* __restrict__ out)
{
    const int rid = blockIdx.x * blockDim.x + threadIdx.x;
    if (rid >= B_ * N_) return;
    float a[E_];
#pragma unroll
    for (int e = 0; e < E_; ++e) a[e] = in[rid * E_ + e];
#pragma unroll
    for (int oo = 0; oo < E_; ++oo) {
        float s = bias[oo];
#pragma unroll
        for (int e = 0; e < E_; ++e) s += a[e] * W[oo * E_ + e];
        out[rid * E_ + oo] = s;
    }
}

// ---------------------------------------------------------------------------
// Kernel 5: cross tail — (attn @ Wo_c^T + bo_c) @ W_out^T + b_out + x3  -> A
// ---------------------------------------------------------------------------
__global__ void k_coordsA(const float* __restrict__ in, const float* __restrict__ Wo,
                          const float* __restrict__ bo, const float* __restrict__ Wout,
                          const float* __restrict__ bout, const float* __restrict__ x3c,
                          float* __restrict__ A)
{
    const int rid = blockIdx.x * blockDim.x + threadIdx.x;
    if (rid >= B_ * N_) return;
    float a[E_];
#pragma unroll
    for (int e = 0; e < E_; ++e) a[e] = in[rid * E_ + e];
    float t12[E_];
#pragma unroll
    for (int oo = 0; oo < E_; ++oo) {
        float s = bo[oo];
#pragma unroll
        for (int e = 0; e < E_; ++e) s += a[e] * Wo[oo * E_ + e];
        t12[oo] = s;
    }
#pragma unroll
    for (int j = 0; j < 3; ++j) {
        float s = bout[j];
#pragma unroll
        for (int oo = 0; oo < E_; ++oo) s += t12[oo] * Wout[j * E_ + oo];
        A[rid * 3 + j] = s + x3c[rid * 3 + j];
    }
}

// ---------------------------------------------------------------------------
// Kernel 6: Kabsch per batch.  H = (Bp-cB)^T (A-cA); R = polar(H) via Newton.
// ---------------------------------------------------------------------------
__device__ __forceinline__ float det3_(const float* X)
{
    return X[0] * (X[4] * X[8] - X[5] * X[7])
         - X[1] * (X[3] * X[8] - X[5] * X[6])
         + X[2] * (X[3] * X[7] - X[4] * X[6]);
}

__global__ void k_kabsch(const float* __restrict__ A, const float* __restrict__ Bp,
                         float* __restrict__ Rt)
{
    __shared__ float acc[15];
    const int b = blockIdx.x, t = threadIdx.x;
    if (t < 15) acc[t] = 0.f;
    __syncthreads();

    float sA[3] = {0.f, 0.f, 0.f}, sB[3] = {0.f, 0.f, 0.f}, M[9];
#pragma unroll
    for (int i = 0; i < 9; ++i) M[i] = 0.f;
    for (int n = t; n < N_; n += 256) {
        const float* ar = &A[(b * N_ + n) * 3];
        const float* br = &Bp[(b * N_ + n) * 3];
#pragma unroll
        for (int i = 0; i < 3; ++i) { sA[i] += ar[i]; sB[i] += br[i]; }
#pragma unroll
        for (int i = 0; i < 3; ++i)
#pragma unroll
            for (int j = 0; j < 3; ++j) M[i * 3 + j] += br[i] * ar[j];
    }
#pragma unroll
    for (int i = 0; i < 3; ++i) { atomicAdd(&acc[i], sA[i]); atomicAdd(&acc[3 + i], sB[i]); }
#pragma unroll
    for (int i = 0; i < 9; ++i) atomicAdd(&acc[6 + i], M[i]);
    __syncthreads();

    if (t == 0) {
        float cA[3], cB[3];
#pragma unroll
        for (int i = 0; i < 3; ++i) { cA[i] = acc[i] * (1.f / N_); cB[i] = acc[3 + i] * (1.f / N_); }
        float X[9];
#pragma unroll
        for (int i = 0; i < 3; ++i)
#pragma unroll
            for (int j = 0; j < 3; ++j)
                X[i * 3 + j] = acc[6 + i * 3 + j] - (float)N_ * cB[i] * cA[j];

        float fn = 0.f;
#pragma unroll
        for (int i = 0; i < 9; ++i) fn += X[i] * X[i];
        fn = sqrtf(fn * (1.f / 3.f)) + 1e-20f;
#pragma unroll
        for (int i = 0; i < 9; ++i) X[i] /= fn;

        for (int it = 0; it < 20; ++it) {
            const float d = det3_(X);
            const float invd = 1.f / d;
            float C[9];
            C[0] =  (X[4] * X[8] - X[5] * X[7]);
            C[1] = -(X[3] * X[8] - X[5] * X[6]);
            C[2] =  (X[3] * X[7] - X[4] * X[6]);
            C[3] = -(X[1] * X[8] - X[2] * X[7]);
            C[4] =  (X[0] * X[8] - X[2] * X[6]);
            C[5] = -(X[0] * X[7] - X[1] * X[6]);
            C[6] =  (X[1] * X[5] - X[2] * X[4]);
            C[7] = -(X[0] * X[5] - X[2] * X[3]);
            C[8] =  (X[0] * X[4] - X[1] * X[3]);
#pragma unroll
            for (int i = 0; i < 9; ++i) X[i] = 0.5f * (X[i] + C[i] * invd);
        }
#pragma unroll
        for (int i = 0; i < 9; ++i) Rt[b * 16 + i] = X[i];
#pragma unroll
        for (int j = 0; j < 3; ++j)
            Rt[b * 16 + 9 + j] =
                cA[j] - (cB[0] * X[0 * 3 + j] + cB[1] * X[1 * 3 + j] + cB[2] * X[2 * 3 + j]);
    }
}

// ---------------------------------------------------------------------------
// Kernel 7: out = x3 @ R + t + ymean3
// ---------------------------------------------------------------------------
__global__ void k_final(const float* __restrict__ x3c, const float* __restrict__ Rt,
                        const float* __restrict__ ymean3, float* __restrict__ out)
{
    const int rid = blockIdx.x * blockDim.x + threadIdx.x;
    if (rid >= B_ * N_) return;
    const int b = rid / N_;
    const float x0 = x3c[rid * 3 + 0], x1 = x3c[rid * 3 + 1], x2 = x3c[rid * 3 + 2];
    const float* R = &Rt[b * 16];
#pragma unroll
    for (int j = 0; j < 3; ++j)
        out[rid * 3 + j] = x0 * R[0 * 3 + j] + x1 * R[1 * 3 + j] + x2 * R[2 * 3 + j]
                         + R[9 + j] + ymean3[b * 3 + j];
}

// ---------------------------------------------------------------------------
extern "C" void kernel_launch(void* const* d_in, const int* in_sizes, int n_in,
                              void* d_out, int out_size, void* d_ws, size_t ws_size,
                              hipStream_t stream)
{
    (void)in_sizes; (void)n_in; (void)out_size; (void)ws_size;
    const float* x_orig = (const float*)d_in[0];
    const float* y_orig = (const float*)d_in[1];
    const float* W_in   = (const float*)d_in[2];
    const float* b_in   = (const float*)d_in[3];
    const float* Wqkv_s = (const float*)d_in[4];
    const float* bqkv_s = (const float*)d_in[5];
    const float* Wo_s   = (const float*)d_in[6];
    const float* bo_s   = (const float*)d_in[7];
    const float* Wqkv_c = (const float*)d_in[8];
    const float* bqkv_c = (const float*)d_in[9];
    const float* Wo_c   = (const float*)d_in[10];
    const float* bo_c   = (const float*)d_in[11];
    const float* W_out  = (const float*)d_in[12];
    const float* b_out  = (const float*)d_in[13];

    float* ws = (float*)d_ws;
    size_t off = 0;
    float* x3c    = ws + off; off += (size_t)B_ * N_ * 3;
    float* Abuf   = ws + off; off += (size_t)B_ * N_ * 3;
    float* ymean3 = ws + off; off += 64;
    float* Rt     = ws + off; off += (size_t)B_ * 16;
    float* xi     = ws + off; off += (size_t)B_ * N_ * E_;
    float* yi     = ws + off; off += (size_t)B_ * N_ * E_;
    float* qb     = ws + off; off += (size_t)B_ * N_ * E_;
    float* kb     = ws + off; off += (size_t)B_ * N_ * E_;
    float* vb     = ws + off; off += (size_t)B_ * N_ * E_;   // (B,H,DH,N) layout
    float* attn   = ws + off; off += (size_t)B_ * N_ * E_;

    const int rows = B_ * N_;
    const int gb   = rows / 256;
    const dim3 fgrid(N_ / 128, H_, B_);

    k_center_project<<<B_, 256, 0, stream>>>(x_orig, y_orig, W_in, b_in, x3c, xi, yi, ymean3);

    // self-attention on xi
    k_qkv<<<gb, 256, 0, stream>>>(xi, Wqkv_s, bqkv_s, qb, kb, vb);
    k_flash<<<fgrid, 256, 0, stream>>>(qb, kb, vb, attn);
    k_proj12<<<gb, 256, 0, stream>>>(attn, Wo_s, bo_s, xi);       // xi <- xi2

    // self-attention on yi
    k_qkv<<<gb, 256, 0, stream>>>(yi, Wqkv_s, bqkv_s, qb, kb, vb);
    k_flash<<<fgrid, 256, 0, stream>>>(qb, kb, vb, attn);
    k_proj12<<<gb, 256, 0, stream>>>(attn, Wo_s, bo_s, yi);       // yi <- yi2

    // cross attention: q from xi2, k/v from yi2
    k_qkv<<<gb, 256, 0, stream>>>(yi, Wqkv_c, bqkv_c, nullptr, kb, vb);
    k_qkv<<<gb, 256, 0, stream>>>(xi, Wqkv_c, bqkv_c, qb, nullptr, nullptr);
    k_flash<<<fgrid, 256, 0, stream>>>(qb, kb, vb, attn);
    k_coordsA<<<gb, 256, 0, stream>>>(attn, Wo_c, bo_c, W_out, b_out, x3c, Abuf);

    // Kabsch + final rigid transform
    k_kabsch<<<B_, 256, 0, stream>>>(Abuf, x3c, Rt);
    k_final<<<gb, 256, 0, stream>>>(x3c, Rt, ymean3, (float*)d_out);
}
